// GNN_41051297415239
// MI455X (gfx1250) — matrix operand order repaired and checked by reading, not compiled, not run on hardware
//
#include <hip/hip_runtime.h>

typedef __attribute__((ext_vector_type(16))) _Float16 v16h;
typedef __attribute__((ext_vector_type(8)))  float    v8f;

#define N_NODES 100000
#define N_EDGES 1600000
#define F_IN    64
#define HID     128
#define OUTF    64

// ---------------------------------------------------------------- utilities
__global__ void zero_f32(float* __restrict__ p, int n) {
  int i = blockIdx.x * blockDim.x + threadIdx.x;
  if (i < n) p[i] = 0.0f;
}

__global__ void combine_bias(const float* __restrict__ ba, const float* __restrict__ bb,
                             float* __restrict__ out, int n) {
  int i = blockIdx.x * blockDim.x + threadIdx.x;
  if (i < n) out[i] = ba[i] + bb[i];
}

// Pre-pack a K x N f32 weight (optionally summed with a second one) into
// WMMA B-fragment-major f16 layout: frag(kb32, ct) -> 32 lanes x 16 halves
// contiguous, so the GEMM loads each lane's v16h as 32 contiguous bytes.
// B-fragment order used here (contiguous k, as in load_a): lane -> N = lane&15, half = lane>>4,
// element j=2v(+1) -> K = kb + half*16 + 2v (+1).
template <int K, int N>
__global__ void pack_b(const float* __restrict__ W0, const float* __restrict__ W1,
                       _Float16* __restrict__ pack) {
  constexpr int NCT = N / 16;
  int i = blockIdx.x * blockDim.x + threadIdx.x;
  if (i >= K * N) return;
  int j    = i & 15;          // element inside lane's v16h
  int lane = (i >> 4) & 31;   // lane
  int frag = i >> 9;          // fragment index = kb32 * NCT + ct
  int kb = (frag / NCT) * 32;
  int ct = frag % NCT;
  int n  = ct * 16 + (lane & 15);
  int h  = lane >> 4;
  int k  = kb + h * 16 + (j & ~1) + (j & 1);
  float w = W0[k * N + n];
  if (W1) w += W1[k * N + n];
  pack[i] = (_Float16)w;
}

// ---------------------------------------------------------------- graph part
__global__ void deg_count(const int* __restrict__ dst, float* __restrict__ deg) {
  int e = blockIdx.x * blockDim.x + threadIdx.x;
  if (e < N_EDGES) atomicAdd(&deg[dst[e]], 1.0f);
}

__global__ void deg_recip(float* __restrict__ deg) {
  int i = blockIdx.x * blockDim.x + threadIdx.x;
  if (i < N_NODES) deg[i] = 1.0f / fmaxf(deg[i], 1.0f);
}

// Scatter-add of source features into destination accumulators.
// One thread = one edge x 4 features (float4 load, 4 f32 atomics).
template <int F>
__global__ void scatter_add(const float* __restrict__ x, const int* __restrict__ src,
                            const int* __restrict__ dst, float* __restrict__ agg) {
  constexpr int Q = F / 4;
  int t = blockIdx.x * blockDim.x + threadIdx.x;
  if (t >= N_EDGES * Q) return;
  int e = t / Q;
  int f = (t - e * Q) * 4;
  int s = src[e];
  int d = dst[e];
  const float4 v = *(const float4*)(x + s * F + f);
  float* o = agg + d * F + f;
  atomicAdd(o + 0, v.x);
  atomicAdd(o + 1, v.y);
  atomicAdd(o + 2, v.z);
  atomicAdd(o + 3, v.w);
}

// ---------------------------------------------------------------- WMMA GEMM
// A-fragment order used here (contiguous k, as in pack_b): lane -> M = lane&15, half = lane>>4.
//   VGPR v in 0..7 : K = half*16 + 2v, +1
__device__ __forceinline__ v16h load_a(const float* __restrict__ A, int rowBase, int K,
                                       int kb, const float* __restrict__ rdeg, int lane) {
  int m = lane & 15, h = lane >> 4;
  int row = rowBase + m;
  float s = rdeg ? rdeg[row] : 1.0f;
  const float* ap = A + row * K + kb;
  v16h a;
#pragma unroll
  for (int v = 0; v < 8; ++v) {
    int k = h * 16 + 2 * v;
    a[2 * v]     = (_Float16)(ap[k] * s);
    a[2 * v + 1] = (_Float16)(ap[k + 1] * s);
  }
  return a;
}

// out = act( (agg .* rdeg) @ Bn  +  self @ Bc  + bias )
// Grid: x = M/16 row tiles; block = 128 threads = 4 waves; each wave owns
// NT = NCOLS/64 column tiles, reusing one A fragment per WMMA pair.
// EXEC is all-1s throughout (M = 6250*16 exactly, no divergence).
template <int KDIM, int NCOLS, bool RELU>
__global__ void sage_gemm(const float* __restrict__ Aagg, const float* __restrict__ rdeg,
                          const float* __restrict__ Aself,
                          const _Float16* __restrict__ Bn, const _Float16* __restrict__ Bc,
                          const float* __restrict__ bias, float* __restrict__ Out) {
  constexpr int NCT = NCOLS / 16;   // total column tiles
  constexpr int NT  = NCT / 4;      // column tiles per wave
  const int lane = threadIdx.x & 31;
  const int wave = threadIdx.x >> 5;
  const int rowBase = blockIdx.x * 16;

  v8f c[NT];
#pragma unroll
  for (int t = 0; t < NT; ++t) c[t] = (v8f){};

#pragma unroll
  for (int kb = 0; kb < KDIM; kb += 32) {
    v16h a = load_a(Aagg, rowBase, KDIM, kb, rdeg, lane);
#pragma unroll
    for (int t = 0; t < NT; ++t) {
      int ct = wave + t * 4;
      v16h b = *(const v16h*)(Bn + ((((kb >> 5) * NCT) + ct) * 32 + lane) * 16);
      c[t] = __builtin_amdgcn_wmma_f32_16x16x32_f16(false, a, false, b, (short)0, c[t],
                                                    false, false);
    }
  }
#pragma unroll
  for (int kb = 0; kb < KDIM; kb += 32) {
    v16h a = load_a(Aself, rowBase, KDIM, kb, nullptr, lane);
#pragma unroll
    for (int t = 0; t < NT; ++t) {
      int ct = wave + t * 4;
      v16h b = *(const v16h*)(Bc + ((((kb >> 5) * NCT) + ct) * 32 + lane) * 16);
      c[t] = __builtin_amdgcn_wmma_f32_16x16x32_f16(false, a, false, b, (short)0, c[t],
                                                    false, false);
    }
  }

  // D layout: lane&15 = N column, VGPR v -> M = v + 8*(lane>>4)
  const int h = lane >> 4;
#pragma unroll
  for (int t = 0; t < NT; ++t) {
    const int n = (wave + t * 4) * 16 + (lane & 15);
    const float bb = bias[n];
#pragma unroll
    for (int v = 0; v < 8; ++v) {
      float val = c[t][v] + bb;
      if (RELU) val = fmaxf(val, 0.0f);
      Out[(rowBase + v + 8 * h) * NCOLS + n] = val;
    }
  }
}

// ---------------------------------------------------------------- launcher
extern "C" void kernel_launch(void* const* d_in, const int* in_sizes, int n_in,
                              void* d_out, int out_size, void* d_ws, size_t ws_size,
                              hipStream_t stream) {
  const float* x   = (const float*)d_in[0];
  const int*   ei  = (const int*)d_in[1];
  const int*   src = ei;
  const int*   dst = ei + N_EDGES;
  const float* W1n = (const float*)d_in[2];
  const float* b1n = (const float*)d_in[3];
  const float* W1r = (const float*)d_in[4];
  const float* Wl1 = (const float*)d_in[5];
  const float* bl1 = (const float*)d_in[6];
  const float* W2n = (const float*)d_in[7];
  const float* b2n = (const float*)d_in[8];
  const float* W2r = (const float*)d_in[9];
  const float* Wl2 = (const float*)d_in[10];
  const float* bl2 = (const float*)d_in[11];

  // workspace carve-out (256B aligned)
  char* ws = (char*)d_ws;
  auto aln = [](size_t o) { return (o + 255) & ~(size_t)255; };
  size_t off = 0;
  float* deg = (float*)(ws + off); off = aln(off + (size_t)N_NODES * 4);
  float* agg = (float*)(ws + off); off = aln(off + (size_t)N_NODES * HID * 4);  // shared L1/L2 accumulator
  float* hbf = (float*)(ws + off); off = aln(off + (size_t)N_NODES * HID * 4);
  _Float16* w1n_p = (_Float16*)(ws + off); off = aln(off + (size_t)F_IN * HID * 2);
  _Float16* wc1_p = (_Float16*)(ws + off); off = aln(off + (size_t)F_IN * HID * 2);
  _Float16* w2n_p = (_Float16*)(ws + off); off = aln(off + (size_t)HID * OUTF * 2);
  _Float16* wc2_p = (_Float16*)(ws + off); off = aln(off + (size_t)HID * OUTF * 2);
  float* bc1 = (float*)(ws + off); off = aln(off + HID * 4);
  float* bc2 = (float*)(ws + off); off = aln(off + OUTF * 4);

  // 1) fold + pack parameters into WMMA B-fragment layout (f16)
  pack_b<F_IN, HID><<<(F_IN * HID + 255) / 256, 256, 0, stream>>>(W1n, nullptr, w1n_p);
  pack_b<F_IN, HID><<<(F_IN * HID + 255) / 256, 256, 0, stream>>>(W1r, Wl1, wc1_p);
  pack_b<HID, OUTF><<<(HID * OUTF + 255) / 256, 256, 0, stream>>>(W2n, nullptr, w2n_p);
  pack_b<HID, OUTF><<<(HID * OUTF + 255) / 256, 256, 0, stream>>>(W2r, Wl2, wc2_p);
  combine_bias<<<1, HID, 0, stream>>>(b1n, bl1, bc1, HID);
  combine_bias<<<1, OUTF, 0, stream>>>(b2n, bl2, bc2, OUTF);

  // 2) degrees + layer-1 scatter-mean numerator
  zero_f32<<<(N_NODES + 255) / 256, 256, 0, stream>>>(deg, N_NODES);
  zero_f32<<<(N_NODES * F_IN + 255) / 256, 256, 0, stream>>>(agg, N_NODES * F_IN);
  deg_count<<<(N_EDGES + 255) / 256, 256, 0, stream>>>(dst, deg);
  scatter_add<F_IN><<<(N_EDGES * (F_IN / 4) + 255) / 256, 256, 0, stream>>>(x, src, dst, agg);
  deg_recip<<<(N_NODES + 255) / 256, 256, 0, stream>>>(deg);

  // 3) h = relu(mean@W1n + x@(W1r+Wl1) + (b1n+bl1))
  sage_gemm<F_IN, HID, true><<<N_NODES / 16, 128, 0, stream>>>(
      agg, deg, x, w1n_p, wc1_p, bc1, hbf);

  // 4) layer-2 scatter-mean numerator (reuse accumulator)
  zero_f32<<<(N_NODES * HID + 255) / 256, 256, 0, stream>>>(agg, N_NODES * HID);
  scatter_add<HID><<<(N_EDGES * (HID / 4) + 255) / 256, 256, 0, stream>>>(hbf, src, dst, agg);

  // 5) out = mean2@W2n + h@(W2r+Wl2) + (b2n+bl2)
  sage_gemm<HID, OUTF, false><<<N_NODES / 16, 128, 0, stream>>>(
      agg, deg, hbf, w2n_p, wc2_p, bc2, (float*)d_out);
}
